// MultiHeadAttention_85074712199466
// MI455X (gfx1250) — matrix operand order repaired and checked by reading, not compiled, not run on hardware
//
#include <hip/hip_runtime.h>
#include <stdint.h>

#define DIM   2048
#define HEADS 16
#define HD    128
#define BATCH 2
#define SEQ   2048
#define N3D   (3 * DIM)
#define NROWS (BATCH * SEQ)   // 4096

typedef __attribute__((ext_vector_type(16))) __bf16 v16bf;
typedef __attribute__((ext_vector_type(8)))  float  v8f;

union FragAB { v16bf v; uint4 q[2]; };

__device__ __forceinline__ unsigned short f2bf(float f) {
  union { float f; unsigned int u; } x; x.f = f;
  unsigned int r = x.u + 0x7FFFu + ((x.u >> 16) & 1u);  // round-to-nearest-even
  return (unsigned short)(r >> 16);
}
__device__ __forceinline__ unsigned int pack2bf(float a, float b) {
  return (unsigned int)f2bf(a) | ((unsigned int)f2bf(b) << 16);
}
__device__ __forceinline__ v8f vzero8() {
  v8f z = {0.f,0.f,0.f,0.f,0.f,0.f,0.f,0.f};
  return z;
}

// ---------------------------------------------------------------------------
// Tensor Data Mover path (gfx1250): 2D bf16 tile -> LDS with row padding.
// This toolchain's builtin takes 6 args: (u32x4, i32x8, i32x4, i32x4, i32x8, cpol).
// Guarded so the cooperative-load fallback is used if the builtin is absent.
// ---------------------------------------------------------------------------
#if defined(__gfx1250__) && __has_builtin(__builtin_amdgcn_tensor_load_to_lds) && \
    __has_builtin(__builtin_amdgcn_s_wait_tensorcnt)
#define USE_TDM 1
typedef unsigned int tdm_u4 __attribute__((ext_vector_type(4)));
typedef int          tdm_i8 __attribute__((ext_vector_type(8)));
typedef int          tdm_i4 __attribute__((ext_vector_type(4)));

// tile_d0 = elements per row, tile_d1 = rows; d0_stride = row pitch (elements)
// pad_int_enc / pad_amt_enc: D# encodings (pad every 2^(1+enc) dwords by enc+1 dwords)
__device__ __forceinline__ void tdm_load_2d_bf16(
    unsigned lds_addr, const void* gaddr,
    unsigned tile_d0, unsigned tile_d1,
    unsigned tensor_d0, unsigned tensor_d1,
    unsigned d0_stride, unsigned pad_int_enc, unsigned pad_amt_enc)
{
  unsigned long long ga = (unsigned long long)(uintptr_t)gaddr;
  tdm_u4 g0;
  g0[0] = 1u;                                            // count=1, no gather
  g0[1] = lds_addr;                                      // LDS byte address
  g0[2] = (unsigned)(ga & 0xffffffffu);                  // global_addr[31:0]
  g0[3] = (unsigned)((ga >> 32) & 0x1ffffffu) | (2u << 30);  // [56:32] | type=2
  tdm_i8 g1;
  g1[0] = (int)((1u << 16) |                             // data_size = 2 bytes
                (1u << 20) |                             // pad_enable
                (pad_int_enc << 22) | (pad_amt_enc << 25));
  g1[1] = (int)((tensor_d0 & 0xffffu) << 16);            // tensor_dim0[15:0]
  g1[2] = (int)((tensor_d0 >> 16) | ((tensor_d1 & 0xffffu) << 16));
  g1[3] = (int)((tensor_d1 >> 16) | (tile_d0 << 16));    // tile_dim0
  g1[4] = (int)tile_d1;                                  // tile_dim1 (tile_dim2=0)
  g1[5] = (int)d0_stride;                                // tensor_dim0_stride[31:0]
  g1[6] = 0;                                             // stride hi / dim1_stride lo
  g1[7] = 0;
  tdm_i4 gz4 = {0, 0, 0, 0};
  tdm_i8 gz8 = {0, 0, 0, 0, 0, 0, 0, 0};
  __builtin_amdgcn_tensor_load_to_lds(g0, g1, gz4, gz4, gz8, 0);
}
#endif

// ---------------------------------------------------------------------------
// Kernel 1: QKV projection GEMM (fp32 in, bf16 WMMA), scatter to
//   Q,K: [b,h,s,hd]   V: [b,h,hd,s] (pre-transposed for attention PV stage)
// grid = (N3D/128, NROWS/128), block = 256 (8 waves)
// ---------------------------------------------------------------------------
__global__ __launch_bounds__(256, 2)
void qkv_proj_kernel(const float* __restrict__ X, const float* __restrict__ W,
                     const float* __restrict__ bias,
                     unsigned short* __restrict__ Qo, unsigned short* __restrict__ Ko,
                     unsigned short* __restrict__ Vo)
{
  __shared__ __align__(16) unsigned short lA[128][40];  // m x k (padded)
  __shared__ __align__(16) unsigned short lB[128][40];  // n x k (W transposed)

  const int n0   = blockIdx.x * 128;
  const int m0   = blockIdx.y * 128;
  const int tid  = threadIdx.x;
  const int lane = tid & 31;
  const int w    = tid >> 5;
  const int wm   = (w & 3) * 32;
  const int wn   = (w >> 2) * 64;
  const int l15  = lane & 15;
  const int hi   = lane >> 4;

  v8f acc[2][4];
  #pragma unroll
  for (int i = 0; i < 2; i++)
    #pragma unroll
    for (int j = 0; j < 4; j++) acc[i][j] = vzero8();

  const int arow = tid >> 1;          // 0..127
  const int akh  = (tid & 1) * 16;    // k half
  const int bk   = tid & 31;          // k row of W tile (wave = one n segment)
  const int bns  = (tid >> 5) * 16;   // n segment

  for (int kt = 0; kt < DIM; kt += 32) {
    { // A tile: fp32 -> bf16
      const float4* src = (const float4*)(X + (size_t)(m0 + arow) * DIM + kt + akh);
      #pragma unroll
      for (int i = 0; i < 4; i++) {
        float4 f = src[i];
        *(unsigned int*)&lA[arow][akh + 4*i    ] = pack2bf(f.x, f.y);
        *(unsigned int*)&lA[arow][akh + 4*i + 2] = pack2bf(f.z, f.w);
      }
    }
    { // W tile, transposed into [n][k]
      const float4* src = (const float4*)(W + (size_t)(kt + bk) * N3D + n0 + bns);
      #pragma unroll
      for (int i = 0; i < 4; i++) {
        float4 f = src[i];
        lB[bns + 4*i    ][bk] = f2bf(f.x);
        lB[bns + 4*i + 1][bk] = f2bf(f.y);
        lB[bns + 4*i + 2][bk] = f2bf(f.z);
        lB[bns + 4*i + 3][bk] = f2bf(f.w);
      }
    }
    __syncthreads();

    FragAB a[2], b[4];
    const int kbA = hi * 8;
    const int kbB = hi * 8;
    #pragma unroll
    for (int i = 0; i < 2; i++) {
      const int m = wm + i*16 + l15;
      a[i].q[0] = *(const uint4*)&lA[m][kbA];
      a[i].q[1] = *(const uint4*)&lA[m][kbA + 16];
    }
    #pragma unroll
    for (int j = 0; j < 4; j++) {
      const int n = wn + j*16 + l15;
      b[j].q[0] = *(const uint4*)&lB[n][kbB];
      b[j].q[1] = *(const uint4*)&lB[n][kbB + 16];
    }
    #pragma unroll
    for (int i = 0; i < 2; i++)
      #pragma unroll
      for (int j = 0; j < 4; j++)
        acc[i][j] = __builtin_amdgcn_wmma_f32_16x16x32_bf16(
            false, a[i].v, false, b[j].v, (short)0, acc[i][j], false, false);
    __syncthreads();
  }

  // Epilogue: bias + scatter into Q/K (row-major) and V (transposed)
  #pragma unroll
  for (int j = 0; j < 4; j++) {
    const int n     = n0 + wn + j*16 + l15;
    const float bj  = bias[n];
    const int h     = n / (3*HD);
    const int c     = n - h * (3*HD);
    const int which = c >> 7;
    const int d     = c & (HD - 1);
    #pragma unroll
    for (int i = 0; i < 2; i++) {
      #pragma unroll
      for (int r = 0; r < 8; r++) {
        const int mrow = m0 + wm + i*16 + r + hi*8;
        const int bidx = mrow >> 11;        // / SEQ
        const int s    = mrow & (SEQ - 1);
        const unsigned short val = f2bf(acc[i][j][r] + bj);
        if (which == 2) {
          Vo[((size_t)(bidx * HEADS + h) * HD + d) * SEQ + s] = val;   // [b,h,d,s]
        } else {
          unsigned short* dst = (which == 0) ? Qo : Ko;
          dst[((size_t)(bidx * HEADS + h) * SEQ + s) * HD + d] = val;  // [b,h,s,d]
        }
      }
    }
  }
}

// ---------------------------------------------------------------------------
// Kernel 2: causal flash attention, bf16 WMMA, f32 online softmax.
// K and V^T tiles staged to LDS by the Tensor Data Mover when available.
// grid = (SEQ/128, BATCH*HEADS), block = 256 (8 waves; wave owns 16 q rows)
// ---------------------------------------------------------------------------
__global__ __launch_bounds__(256, 1)
void attn_kernel(const unsigned short* __restrict__ Q,
                 const unsigned short* __restrict__ K,
                 const unsigned short* __restrict__ V,   // [b,h,d,s] transposed
                 unsigned short* __restrict__ O)         // [B*S][DIM] bf16
{
  __shared__ __align__(16) unsigned short lK [64][136];   // kv x d (pad 8)
  __shared__ __align__(16) unsigned short lVt[128][72];   // d  x kv (pad 8)
  __shared__ __align__(16) unsigned short lP [8][16][72]; // per-wave P relayout

  const int qb   = blockIdx.x * 128;
  const int bh   = blockIdx.y;
  const int b    = bh >> 4;
  const int h    = bh & 15;
  const int tid  = threadIdx.x;
  const int lane = tid & 31;
  const int w    = tid >> 5;
  const int l15  = lane & 15;
  const int hi   = lane >> 4;

  const unsigned short* Qb  = Q + (size_t)bh * SEQ * HD;
  const unsigned short* Kb  = K + (size_t)bh * SEQ * HD;
  const unsigned short* Vtb = V + (size_t)bh * HD * SEQ;

  // Load this wave's Q fragments once (16 rows, K=128 -> 4 frags)
  FragAB qf[4];
  {
    const int m  = qb + w*16 + l15;
    const int kb = hi * 8;
    #pragma unroll
    for (int f = 0; f < 4; f++) {
      qf[f].q[0] = *(const uint4*)&Qb[(size_t)m * HD + f*32 + kb];
      qf[f].q[1] = *(const uint4*)&Qb[(size_t)m * HD + f*32 + kb + 16];
    }
  }

  v8f accO[8];
  #pragma unroll
  for (int t = 0; t < 8; t++) accO[t] = vzero8();
  float mrow[8], lrow[8];
  #pragma unroll
  for (int r = 0; r < 8; r++) { mrow[r] = -3.0e38f; lrow[r] = 0.f; }

  const float scale = 0.08838834764831845f;  // 1/sqrt(HD)
  const int kvend = qb + 128;                // causal: only blocks on/below diagonal

  for (int kv0 = 0; kv0 < kvend; kv0 += 64) {
#ifdef USE_TDM
    // One wave issues each 2D tile DMA; D# padding produces the LDS pitch.
    if (w == 0) {
      // K tile: 64 rows x 128 elems, row pitch HD; pad 4 dwords per 64 dwords
      tdm_load_2d_bf16((unsigned)(uintptr_t)&lK[0][0],
                       Kb + (size_t)kv0 * HD,
                       HD, 64, HD, SEQ, HD, 5u, 3u);
      __builtin_amdgcn_s_wait_tensorcnt(0);
    } else if (w == 1) {
      // V^T tile: 128 rows x 64 elems, row pitch SEQ; pad 4 dwords per 16 dwords
      tdm_load_2d_bf16((unsigned)(uintptr_t)&lVt[0][0],
                       Vtb + kv0,
                       64, HD, SEQ, HD, SEQ, 3u, 3u);
      __builtin_amdgcn_s_wait_tensorcnt(0);
    }
#else
    { // coop load K tile (row-major)
      const int kv = tid >> 2;
      const int ds = (tid & 3) * 32;
      const uint4* src = (const uint4*)&Kb[(size_t)(kv0 + kv) * HD + ds];
      uint4* dq = (uint4*)&lK[kv][ds];
      #pragma unroll
      for (int i = 0; i < 4; i++) dq[i] = src[i];
    }
    { // coop load V^T tile (already transposed in global)
      const int d  = tid >> 1;
      const int ks = (tid & 1) * 32;
      const uint4* src = (const uint4*)&Vtb[(size_t)d * SEQ + kv0 + ks];
      uint4* dq = (uint4*)&lVt[d][ks];
      #pragma unroll
      for (int i = 0; i < 4; i++) dq[i] = src[i];
    }
#endif
    __syncthreads();

    // S = Q * K^T : 4 n-tiles x 4 k-steps = 16 WMMA
    v8f sc[4];
    #pragma unroll
    for (int nt = 0; nt < 4; nt++) sc[nt] = vzero8();
    #pragma unroll
    for (int nt = 0; nt < 4; nt++) {
      const int kvn = nt*16 + l15;
      const int kb2 = hi * 8;
      #pragma unroll
      for (int kf = 0; kf < 4; kf++) {
        FragAB bf;
        bf.q[0] = *(const uint4*)&lK[kvn][kf*32 + kb2];
        bf.q[1] = *(const uint4*)&lK[kvn][kf*32 + kb2 + 16];
        sc[nt] = __builtin_amdgcn_wmma_f32_16x16x32_bf16(
            false, qf[kf].v, false, bf.v, (short)0, sc[nt], false, false);
      }
    }

    // scale + causal mask (skipped for blocks strictly below the diagonal)
    const bool needMask = (kv0 + 64) > (qb + w*16);  // wave-uniform
    #pragma unroll
    for (int r = 0; r < 8; r++) {
      const int qrow = qb + w*16 + r + hi*8;
      float mx = -3.0e38f;
      #pragma unroll
      for (int nt = 0; nt < 4; nt++) {
        const int kcol = kv0 + nt*16 + l15;
        float v = sc[nt][r] * scale;
        if (needMask && (kcol > qrow)) v = -3.0e38f;
        sc[nt][r] = v;
        mx = fmaxf(mx, v);
      }
      #pragma unroll
      for (int off = 1; off < 16; off <<= 1)
        mx = fmaxf(mx, __shfl_xor(mx, off, 32));
      const float mnew = fmaxf(mrow[r], mx);
      const float corr = __expf(mrow[r] - mnew);
      float rs = 0.f;
      #pragma unroll
      for (int nt = 0; nt < 4; nt++) {
        float p = __expf(sc[nt][r] - mnew);
        sc[nt][r] = p;
        rs += p;
      }
      #pragma unroll
      for (int off = 1; off < 16; off <<= 1)
        rs += __shfl_xor(rs, off, 32);
      lrow[r] = lrow[r] * corr + rs;
      mrow[r] = mnew;
      #pragma unroll
      for (int t = 0; t < 8; t++) accO[t][r] *= corr;
    }

    // relayout P (C-layout f32 -> A-layout bf16) through per-wave LDS
    #pragma unroll
    for (int nt = 0; nt < 4; nt++)
      #pragma unroll
      for (int r = 0; r < 8; r++)
        lP[w][r + hi*8][nt*16 + l15] = f2bf(sc[nt][r]);

    FragAB pf[2];
    {
      const int kb = hi * 8;
      #pragma unroll
      for (int f = 0; f < 2; f++) {
        pf[f].q[0] = *(const uint4*)&lP[w][l15][f*32 + kb];
        pf[f].q[1] = *(const uint4*)&lP[w][l15][f*32 + kb + 16];
      }
    }
    // O += P * V : 8 d-tiles x 2 k-steps = 16 WMMA
    #pragma unroll
    for (int dt = 0; dt < 8; dt++) {
      const int d   = dt*16 + l15;
      const int kb2 = hi * 8;
      #pragma unroll
      for (int f = 0; f < 2; f++) {
        FragAB bf;
        bf.q[0] = *(const uint4*)&lVt[d][f*32 + kb2];
        bf.q[1] = *(const uint4*)&lVt[d][f*32 + kb2 + 16];
        accO[dt] = __builtin_amdgcn_wmma_f32_16x16x32_bf16(
            false, pf[f].v, false, bf.v, (short)0, accO[dt], false, false);
      }
    }
    __syncthreads();
  }

  // finalize: O /= l, store bf16 into [b, s, h*HD + d] layout
  #pragma unroll
  for (int r = 0; r < 8; r++) {
    const float inv = 1.0f / lrow[r];
    const int srow  = qb + w*16 + r + hi*8;
    unsigned short* dst = O + ((size_t)(b * SEQ + srow)) * DIM + h * HD;
    #pragma unroll
    for (int dt = 0; dt < 8; dt++)
      dst[dt*16 + l15] = f2bf(accO[dt][r] * inv);
  }
}

// ---------------------------------------------------------------------------
// Kernel 3: output projection GEMM (bf16 A from ws, fp32 W/out + bias)
// grid = (DIM/128, NROWS/128), block = 256
// ---------------------------------------------------------------------------
__global__ __launch_bounds__(256, 2)
void out_proj_kernel(const unsigned short* __restrict__ A,  // [NROWS][DIM] bf16
                     const float* __restrict__ W,           // [DIM][DIM]
                     const float* __restrict__ bias,
                     float* __restrict__ out)
{
  __shared__ __align__(16) unsigned short lA[128][40];
  __shared__ __align__(16) unsigned short lB[128][40];

  const int n0   = blockIdx.x * 128;
  const int m0   = blockIdx.y * 128;
  const int tid  = threadIdx.x;
  const int lane = tid & 31;
  const int w    = tid >> 5;
  const int wm   = (w & 3) * 32;
  const int wn   = (w >> 2) * 64;
  const int l15  = lane & 15;
  const int hi   = lane >> 4;

  v8f acc[2][4];
  #pragma unroll
  for (int i = 0; i < 2; i++)
    #pragma unroll
    for (int j = 0; j < 4; j++) acc[i][j] = vzero8();

  const int arow = tid >> 1;
  const int akh  = (tid & 1) * 16;
  const int bk   = tid & 31;
  const int bns  = (tid >> 5) * 16;

  for (int kt = 0; kt < DIM; kt += 32) {
    { // A tile: straight bf16 copy
      const uint4* src = (const uint4*)&A[(size_t)(m0 + arow) * DIM + kt + akh];
      *(uint4*)&lA[arow][akh    ] = src[0];
      *(uint4*)&lA[arow][akh + 8] = src[1];
    }
    { // W tile transposed
      const float4* src = (const float4*)(W + (size_t)(kt + bk) * DIM + n0 + bns);
      #pragma unroll
      for (int i = 0; i < 4; i++) {
        float4 f = src[i];
        lB[bns + 4*i    ][bk] = f2bf(f.x);
        lB[bns + 4*i + 1][bk] = f2bf(f.y);
        lB[bns + 4*i + 2][bk] = f2bf(f.z);
        lB[bns + 4*i + 3][bk] = f2bf(f.w);
      }
    }
    __syncthreads();

    FragAB a[2], b[4];
    const int kbA = hi * 8;
    const int kbB = hi * 8;
    #pragma unroll
    for (int i = 0; i < 2; i++) {
      const int m = wm + i*16 + l15;
      a[i].q[0] = *(const uint4*)&lA[m][kbA];
      a[i].q[1] = *(const uint4*)&lA[m][kbA + 16];
    }
    #pragma unroll
    for (int j = 0; j < 4; j++) {
      const int n = wn + j*16 + l15;
      b[j].q[0] = *(const uint4*)&lB[n][kbB];
      b[j].q[1] = *(const uint4*)&lB[n][kbB + 16];
    }
    #pragma unroll
    for (int i = 0; i < 2; i++)
      #pragma unroll
      for (int j = 0; j < 4; j++)
        acc[i][j] = __builtin_amdgcn_wmma_f32_16x16x32_bf16(
            false, a[i].v, false, b[j].v, (short)0, acc[i][j], false, false);
    __syncthreads();
  }

  #pragma unroll
  for (int j = 0; j < 4; j++) {
    const int n    = n0 + wn + j*16 + l15;
    const float bj = bias[n];
    #pragma unroll
    for (int i = 0; i < 2; i++) {
      #pragma unroll
      for (int r = 0; r < 8; r++) {
        const int mrow = m0 + wm + i*16 + r + hi*8;
        out[(size_t)mrow * DIM + n] = acc[i][j][r] + bj;
      }
    }
  }
}

// ---------------------------------------------------------------------------
extern "C" void kernel_launch(void* const* d_in, const int* in_sizes, int n_in,
                              void* d_out, int out_size, void* d_ws, size_t ws_size,
                              hipStream_t stream) {
  (void)in_sizes; (void)n_in; (void)out_size; (void)ws_size;
  const float* x     = (const float*)d_in[0];
  // d_in[1] = causal mask (bool) — causality is computed analytically in-kernel
  const float* W_qkv = (const float*)d_in[2];
  const float* b_qkv = (const float*)d_in[3];
  const float* W_out = (const float*)d_in[4];
  const float* b_out = (const float*)d_in[5];

  unsigned short* ws = (unsigned short*)d_ws;
  const size_t perT = (size_t)BATCH * HEADS * SEQ * HD;  // 8Mi bf16 elements
  unsigned short* Qb = ws;
  unsigned short* Kb = ws + perT;
  unsigned short* Vb = ws + 2 * perT;   // stored transposed [b,h,d,s]
  unsigned short* Ob = ws + 3 * perT;   // [B*S][DIM] bf16

  qkv_proj_kernel<<<dim3(N3D / 128, NROWS / 128), 256, 0, stream>>>(
      x, W_qkv, b_qkv, Qb, Kb, Vb);
  attn_kernel<<<dim3(SEQ / 128, BATCH * HEADS), 256, 0, stream>>>(
      Qb, Kb, Vb, Ob);
  out_proj_kernel<<<dim3(DIM / 128, NROWS / 128), 256, 0, stream>>>(
      Ob, W_out, b_out, (float*)d_out);
}
